// Mymodel_GAT_cov_nobn_89051851915804
// MI455X (gfx1250) — matrix operand order repaired and checked by reading, not compiled, not run on hardware
//
#include <hip/hip_runtime.h>
#include <hip/hip_bf16.h>

// ---------------- CDNA5 WMMA types ----------------
typedef __attribute__((ext_vector_type(16))) __bf16       v16bf;
typedef __attribute__((ext_vector_type(8)))  float        v8f;
typedef __attribute__((ext_vector_type(4)))  unsigned int u32x4;
typedef __attribute__((ext_vector_type(4)))  int          v4i;

union FragBF {
  v16bf v;
  u32x4 u[2];
};

#define AS1 __attribute__((address_space(1)))
#define AS3 __attribute__((address_space(3)))

#if __has_builtin(__builtin_amdgcn_global_load_async_to_lds_b128)
#define HAVE_ASYNC_LDS 1
#endif

__device__ __forceinline__ void wait_async0()
{
#if __has_builtin(__builtin_amdgcn_s_wait_asynccnt)
  __builtin_amdgcn_s_wait_asynccnt(0);
#else
  asm volatile("s_wait_asynccnt 0" ::: "memory");
#endif
}

// Model constants
#define HEADS   4
#define HID     610
#define ND      2440   // HEADS*HID
#define NDP     2448   // padded to 16 (ldc of h)
#define WROWS   2560   // W rows padded to block-N (128) multiple
#define KP2     640    // HID padded to 32 for K dim
#define KER     50
#define L1LEN   561    // HID - KER + 1
#define L2LEN   512
#define NOUT    86
#define BK      32     // K step

// ======================================================================
// bf16 WMMA GEMM: C[M, Nc] = A[M, Kp](bf16, row major) * B[Nc, Kp](bf16)^T
// Block = 256 threads (8 waves, 2Mx4N). Block tile 64(M) x 128(N).
// Wave tile 32x32 = 2x2 WMMA accumulators.
// DOUBLE-BUFFERED LDS staging: async loads (ASYNCcnt) of K-slice k+32
// overlap the 4 WMMAs on slice k; s_wait_asynccnt sits BEHIND the math.
// Grid must cover PADDED rows: grid.y = rowsP(A)/64, grid.x = rowsP(B)/128.
// ======================================================================
__global__ __launch_bounds__(256)
void wmma_gemm_bf16(const __bf16* __restrict__ A,
                    const __bf16* __restrict__ B,
                    float* __restrict__ C,
                    const float* __restrict__ bias,
                    int M, int Nc, int Kp, int ldc)
{
  __shared__ __bf16 lA[2][64 * BK];    // 2 x 4 KB
  __shared__ __bf16 lB[2][128 * BK];   // 2 x 8 KB

  const int tid  = threadIdx.x;
  const int wave = tid >> 5;
  const int lane = tid & 31;
  const int wm   = wave >> 2;       // 0..1
  const int wn   = wave & 3;        // 0..3
  const int half = lane >> 4;
  const int r    = lane & 15;

  // --- staging assignment (contiguous 16B chunks) ---
  // A tile 64x32 bf16 = 256 chunks -> 1/thread ; B tile 128x32 = 512 -> 2/thread
  const int ar = tid >> 2;
  const int ac = (tid & 3) * 8;
  const int br = tid >> 2;          // + 0 / + 64
  const int bc = (tid & 3) * 8;

  const __bf16* gA  = A + ((size_t)blockIdx.y * 64 + ar) * Kp + ac;
  const __bf16* gB0 = B + ((size_t)blockIdx.x * 128 + br) * Kp + bc;
  const __bf16* gB1 = B + ((size_t)blockIdx.x * 128 + br + 64) * Kp + bc;
  const int sAo  = ar * BK + ac;
  const int sB0o = br * BK + bc;
  const int sB1o = (br + 64) * BK + bc;

  auto stage = [&](int buf, int k) {
#ifdef HAVE_ASYNC_LDS
    __builtin_amdgcn_global_load_async_to_lds_b128(
        (AS1 v4i*)(gA + k),  (AS3 v4i*)(&lA[buf][sAo]),  0, 0);
    __builtin_amdgcn_global_load_async_to_lds_b128(
        (AS1 v4i*)(gB0 + k), (AS3 v4i*)(&lB[buf][sB0o]), 0, 0);
    __builtin_amdgcn_global_load_async_to_lds_b128(
        (AS1 v4i*)(gB1 + k), (AS3 v4i*)(&lB[buf][sB1o]), 0, 0);
#else
    *(u32x4*)&lA[buf][sAo]  = *(const u32x4*)(gA + k);
    *(u32x4*)&lB[buf][sB0o] = *(const u32x4*)(gB0 + k);
    *(u32x4*)&lB[buf][sB1o] = *(const u32x4*)(gB1 + k);
#endif
  };

  v8f acc[2][2] = {};

  // prologue: fill buffer 0 with K-slice 0
  stage(0, 0);
#ifdef HAVE_ASYNC_LDS
  wait_async0();
#endif
  __syncthreads();

  int cur = 0;
  for (int k = 0; k < Kp; k += BK) {
    const int nxt = cur ^ 1;
    if (k + BK < Kp) stage(nxt, k + BK);   // prefetch next slice (overlaps WMMA)

    // --- fragments from LDS (ISA 7.12.2 16-bit layouts) ---
    FragBF a0, a1, b0, b1;
    {
      // A 16x32: lane half0 -> K {0..7}u{16..23}; half1 -> {8..15}u{24..31}
      const __bf16* p0 = &lA[cur][(wm * 32 + r) * BK] + half * 8;
      const __bf16* p1 = &lA[cur][(wm * 32 + 16 + r) * BK] + half * 8;
      a0.u[0] = *(const u32x4*)(p0);
      a0.u[1] = *(const u32x4*)(p0 + 16);
      a1.u[0] = *(const u32x4*)(p1);
      a1.u[1] = *(const u32x4*)(p1 + 16);
      // B 32x16: lane r = column; half0 -> K {0..7}u{16..23}; half1 -> {8..15}u{24..31}
      const __bf16* q0 = &lB[cur][(wn * 32 + r) * BK] + half * 8;
      const __bf16* q1 = &lB[cur][(wn * 32 + 16 + r) * BK] + half * 8;
      b0.u[0] = *(const u32x4*)(q0);
      b0.u[1] = *(const u32x4*)(q0 + 16);
      b1.u[0] = *(const u32x4*)(q1);
      b1.u[1] = *(const u32x4*)(q1 + 16);
    }

    // (neg_a, A, neg_b, B, c_mod, C, reuse_a, reuse_b)
    acc[0][0] = __builtin_amdgcn_wmma_f32_16x16x32_bf16(false, a0.v, false, b0.v,
                                                        (short)0, acc[0][0], false, false);
    acc[0][1] = __builtin_amdgcn_wmma_f32_16x16x32_bf16(false, a0.v, false, b1.v,
                                                        (short)0, acc[0][1], false, false);
    acc[1][0] = __builtin_amdgcn_wmma_f32_16x16x32_bf16(false, a1.v, false, b0.v,
                                                        (short)0, acc[1][0], false, false);
    acc[1][1] = __builtin_amdgcn_wmma_f32_16x16x32_bf16(false, a1.v, false, b1.v,
                                                        (short)0, acc[1][1], false, false);

#ifdef HAVE_ASYNC_LDS
    wait_async0();          // prefetch landed (behind the math)
#endif
    __syncthreads();
    cur = nxt;
  }

  // C/D 16x16 f32 layout: VGPR i: lanes 0-15 -> M=i, lanes 16-31 -> M=8+i
#pragma unroll
  for (int mt = 0; mt < 2; ++mt) {
    const int tM = blockIdx.y * 4 + wm * 2 + mt;
    if (tM * 16 >= M) continue;
    const int row0 = tM * 16 + half * 8;
#pragma unroll
    for (int nt = 0; nt < 2; ++nt) {
      const int col = (blockIdx.x * 8 + wn * 2 + nt) * 16 + r;
      if (col >= Nc) continue;
      const float bv = bias ? bias[col] : 0.0f;
#pragma unroll
      for (int i = 0; i < 8; ++i)
        C[(size_t)(row0 + i) * ldc + col] = acc[mt][nt][i] + bv;
    }
  }
}

// ======================================================================
// f32 -> bf16 with zero padding of both dims (keeps WMMA tiles guard-free)
// ======================================================================
__global__ void cvt_pad_bf16(const float* __restrict__ src, __bf16* __restrict__ dst,
                             int rows, int cols, int colsP, long long total)
{
  long long i = (long long)blockIdx.x * blockDim.x + threadIdx.x;
  if (i >= total) return;
  int r = (int)(i / colsP);
  int c = (int)(i % colsP);
  float v = (r < rows && c < cols) ? src[(size_t)r * cols + c] : 0.0f;
  dst[i] = (__bf16)v;
}

__global__ void fill_f32(float* __restrict__ p, float v, long long n)
{
  long long i = (long long)blockIdx.x * blockDim.x + threadIdx.x;
  if (i < n) p[i] = v;
}

__global__ void fill_bf16(__bf16* __restrict__ p, long long n)
{
  long long i = (long long)blockIdx.x * blockDim.x + threadIdx.x;
  if (i < n) p[i] = (__bf16)0.0f;
}

// ======================================================================
// Attention scores: a_src[n,h] = <h[n,h,:], att_src[h,:]>, same for dst
// ======================================================================
__global__ void att_scores(const float* __restrict__ h,
                           const float* __restrict__ att_s,
                           const float* __restrict__ att_d,
                           float* __restrict__ a_src, float* __restrict__ a_dst,
                           int N, int ldh)
{
  int i = blockIdx.x * blockDim.x + threadIdx.x;
  if (i >= N * HEADS) return;
  int n = i >> 2, hd = i & 3;
  const float* hp = h + (size_t)n * ldh + hd * HID;
  const float* sp = att_s + hd * HID;
  const float* dp = att_d + hd * HID;
  float s = 0.f, t = 0.f;
  for (int d = 0; d < HID; ++d) { float v = hp[d]; s += v * sp[d]; t += v * dp[d]; }
  a_src[i] = s;
  a_dst[i] = t;
}

// Signed float atomic max via int/uint monotone mapping (init must be -inf)
__device__ __forceinline__ void atomicMaxF(float* addr, float val)
{
  if (val >= 0.0f) atomicMax((int*)addr, __float_as_int(val));
  else             atomicMin((unsigned int*)addr, __float_as_uint(val));
}

__global__ void edge_alpha(const int* __restrict__ src, const int* __restrict__ dst,
                           const float* __restrict__ a_src, const float* __restrict__ a_dst,
                           float* __restrict__ alpha, float* __restrict__ m, int E)
{
  int i = blockIdx.x * blockDim.x + threadIdx.x;
  if (i >= E * HEADS) return;
  int e = i >> 2, hd = i & 3;
  float al = a_src[src[e] * HEADS + hd] + a_dst[dst[e] * HEADS + hd];
  al = al > 0.f ? al : 0.2f * al;          // leaky relu
  alpha[i] = al;
  atomicMaxF(&m[dst[e] * HEADS + hd], al);
}

__global__ void edge_exp(const int* __restrict__ dst,
                         float* __restrict__ alpha, const float* __restrict__ m,
                         float* __restrict__ denom, int E)
{
  int i = blockIdx.x * blockDim.x + threadIdx.x;
  if (i >= E * HEADS) return;
  int e = i >> 2, hd = i & 3;
  float ev = __expf(alpha[i] - m[dst[e] * HEADS + hd]);
  alpha[i] = ev;
  atomicAdd(&denom[dst[e] * HEADS + hd], ev);
}

// One block per edge: agg[dst] += coef[h] * h[src]
__global__ __launch_bounds__(256)
void edge_aggregate(const int* __restrict__ src, const int* __restrict__ dst,
                    const float* __restrict__ h, const float* __restrict__ eval,
                    const float* __restrict__ denom, float* __restrict__ agg, int ldh)
{
  __shared__ float coef[HEADS];
  int e = blockIdx.x;
  int s = src[e], dd = dst[e];
  if (threadIdx.x < HEADS)
    coef[threadIdx.x] = eval[e * HEADS + threadIdx.x]
                      / (denom[dd * HEADS + threadIdx.x] + 1e-16f);
  __syncthreads();
  const float* hp = h + (size_t)s * ldh;     // heads contiguous in first ND cols
  float* op = agg + (size_t)dd * ND;
  for (int j = threadIdx.x; j < ND; j += blockDim.x) {
    int hd = j / HID;
    atomicAdd(op + j, coef[hd] * hp[j]);
  }
}

// relu(mean over heads + bias) -> f32 [N,HID] and zero-padded bf16 [rowsP,KP2]
__global__ void finalize_layer(const float* __restrict__ agg, const float* __restrict__ bias,
                               float* __restrict__ xf, __bf16* __restrict__ xb,
                               int N, int colsP, long long total)
{
  long long i = (long long)blockIdx.x * blockDim.x + threadIdx.x;
  if (i >= total) return;
  int n = (int)(i / colsP), d = (int)(i % colsP);
  float v = 0.f;
  if (n < N && d < HID) {
    const float* ap = agg + (size_t)n * ND + d;
    v = 0.25f * (ap[0] + ap[HID] + ap[2 * HID] + ap[3 * HID]) + bias[d];
    v = v > 0.f ? v : 0.f;
    xf[(size_t)n * HID + d] = v;
  }
  xb[i] = (__bf16)v;
}

// ======================================================================
// Per-edge conv stack: [2,610] conv(2x50) -> relu -> [561] conv(50) -> relu
// One block per edge, everything staged in LDS. Output bf16 for final GEMM.
// ======================================================================
__global__ __launch_bounds__(256)
void edge_conv(const float* __restrict__ x2,
               const int* __restrict__ src, const int* __restrict__ dst,
               const float* __restrict__ w4, const float* __restrict__ b4,
               const float* __restrict__ w6, const float* __restrict__ b6,
               __bf16* __restrict__ A3)
{
  __shared__ float xs[HID], xd[HID], y1[L1LEN], k4[2 * KER], k6[KER];
  int e = blockIdx.x;
  int t = threadIdx.x;
  int s = src[e], dd = dst[e];
  for (int j = t; j < HID; j += 256) {
    xs[j] = x2[(size_t)s * HID + j];
    xd[j] = x2[(size_t)dd * HID + j];
  }
  for (int j = t; j < 2 * KER; j += 256) k4[j] = w4[j];
  for (int j = t; j < KER;     j += 256) k6[j] = w6[j];
  __syncthreads();
  const float bb4 = b4[0], bb6 = b6[0];
  for (int j = t; j < L1LEN; j += 256) {
    float acc = bb4;
#pragma unroll 10
    for (int q = 0; q < KER; ++q)
      acc += k4[q] * xs[j + q] + k4[KER + q] * xd[j + q];
    y1[j] = acc > 0.f ? acc : 0.f;
  }
  __syncthreads();
  for (int c = t; c < L2LEN; c += 256) {
    float acc = bb6;
#pragma unroll 10
    for (int q = 0; q < KER; ++q)
      acc += k6[q] * y1[c + q];
    acc = acc > 0.f ? acc : 0.f;
    A3[(size_t)e * L2LEN + c] = (__bf16)acc;
  }
}

// ======================================================================
// Host side
// ======================================================================
static inline int cdiv(long long a, long long b) { return (int)((a + b - 1) / b); }
static inline long long padto(long long v, long long m) { return ((v + m - 1) / m) * m; }

extern "C" void kernel_launch(void* const* d_in, const int* in_sizes, int n_in,
                              void* d_out, int out_size, void* d_ws, size_t ws_size,
                              hipStream_t stream)
{
  const float* x   = (const float*)d_in[0];
  const int*   ei  = (const int*)d_in[1];
  const float* w0  = (const float*)d_in[2];
  const float* b0  = (const float*)d_in[3];
  const float* as0 = (const float*)d_in[4];
  const float* ad0 = (const float*)d_in[5];
  const float* w2  = (const float*)d_in[6];
  const float* b2  = (const float*)d_in[7];
  const float* as1 = (const float*)d_in[8];
  const float* ad1 = (const float*)d_in[9];
  const float* w4  = (const float*)d_in[10];
  const float* b4  = (const float*)d_in[11];
  const float* w6  = (const float*)d_in[12];
  const float* b6  = (const float*)d_in[13];
  const float* w8  = (const float*)d_in[14];
  const float* b8  = (const float*)d_in[15];

  const int FIN = 128;
  const int N = in_sizes[0] / FIN;
  const int E = in_sizes[1] / 2;
  const int* src = ei;
  const int* dst = ei + E;

  const long long NP  = padto(N, 64);    // A-rows pad for block tile
  const long long EP  = padto(E, 64);
  const long long W3R = 128;             // w8 rows padded to 128

  // --- workspace bump allocator (256B aligned) ---
  char* ws = (char*)d_ws;
  size_t off = 0;
  auto alloc = [&](size_t bytes) -> void* {
    off = (off + 255) & ~(size_t)255;
    void* p = ws + off;
    off += bytes;
    return p;
  };
  __bf16* xb    = (__bf16*)alloc((size_t)NP * FIN * 2);
  __bf16* w0b   = (__bf16*)alloc((size_t)WROWS * FIN * 2);
  __bf16* w2b   = (__bf16*)alloc((size_t)WROWS * KP2 * 2);
  __bf16* w8b   = (__bf16*)alloc((size_t)W3R * L2LEN * 2);
  float*  h     = (float*)alloc((size_t)N * NDP * 4);
  float*  agg   = (float*)alloc((size_t)N * ND * 4);
  float*  a_src = (float*)alloc((size_t)N * HEADS * 4);
  float*  a_dst = (float*)alloc((size_t)N * HEADS * 4);
  float*  mbuf  = (float*)alloc((size_t)N * HEADS * 4);
  float*  dbuf  = (float*)alloc((size_t)N * HEADS * 4);
  float*  alpha = (float*)alloc((size_t)E * HEADS * 4);
  float*  xf    = (float*)alloc((size_t)N * HID * 4);
  __bf16* x1b   = (__bf16*)alloc((size_t)NP * KP2 * 2);
  __bf16* A3    = (__bf16*)alloc((size_t)EP * L2LEN * 2);

  const int T = 256;
  (void)n_in; (void)out_size; (void)ws_size;

  // --- weight / input conversions (zero-fill pads) ---
  {
    long long tot = NP * FIN;
    cvt_pad_bf16<<<cdiv(tot, T), T, 0, stream>>>(x, xb, N, FIN, FIN, tot);
  }
  {
    long long tot = (long long)WROWS * FIN;
    cvt_pad_bf16<<<cdiv(tot, T), T, 0, stream>>>(w0, w0b, ND, FIN, FIN, tot);
  }
  {
    long long tot = (long long)WROWS * KP2;
    cvt_pad_bf16<<<cdiv(tot, T), T, 0, stream>>>(w2, w2b, ND, HID, KP2, tot);
  }
  {
    long long tot = W3R * L2LEN;
    cvt_pad_bf16<<<cdiv(tot, T), T, 0, stream>>>(w8, w8b, NOUT, L2LEN, L2LEN, tot);
  }
  {
    long long pad = (EP - E) * L2LEN;
    if (pad > 0)
      fill_bf16<<<cdiv(pad, T), T, 0, stream>>>(A3 + (size_t)E * L2LEN, pad);
  }

  // --- one GAT layer ---
  auto gat = [&](const __bf16* Ain, int Kp, const __bf16* Wb,
                 const float* atts, const float* attd, const float* bias,
                 float* xout_f32, __bf16* xout_bf16) {
    dim3 g((unsigned)(WROWS / 128), (unsigned)(NP / 64));
    wmma_gemm_bf16<<<g, 256, 0, stream>>>(Ain, Wb, h, nullptr, N, ND, Kp, NDP);

    att_scores<<<cdiv((long long)N * HEADS, T), T, 0, stream>>>(
        h, atts, attd, a_src, a_dst, N, NDP);

    fill_f32<<<cdiv((long long)N * HEADS, T), T, 0, stream>>>(
        mbuf, -__builtin_inff(), (long long)N * HEADS);
    fill_f32<<<cdiv((long long)N * HEADS, T), T, 0, stream>>>(
        dbuf, 0.0f, (long long)N * HEADS);
    fill_f32<<<cdiv((long long)N * ND, T), T, 0, stream>>>(
        agg, 0.0f, (long long)N * ND);

    edge_alpha<<<cdiv((long long)E * HEADS, T), T, 0, stream>>>(
        src, dst, a_src, a_dst, alpha, mbuf, E);
    edge_exp<<<cdiv((long long)E * HEADS, T), T, 0, stream>>>(
        dst, alpha, mbuf, dbuf, E);
    edge_aggregate<<<E, 256, 0, stream>>>(src, dst, h, alpha, dbuf, agg, NDP);

    long long tot = NP * KP2;
    finalize_layer<<<cdiv(tot, T), T, 0, stream>>>(
        agg, bias, xout_f32, xout_bf16, N, KP2, tot);
  };

  // Layer 1: x(bf16, K=128) -> x1 (bf16, K padded to 640)
  gat(xb, FIN, w0b, as0, ad0, b0, xf, x1b);
  // Layer 2: x1 -> x2 (f32 in xf; bf16 copy recycled into x1b, unused after)
  gat(x1b, KP2, w2b, as1, ad1, b2, xf, x1b);

  // Per-edge conv stack -> A3 [E, 512] bf16
  edge_conv<<<E, 256, 0, stream>>>(xf, src, dst, w4, b4, w6, b6, A3);

  // Final linear: [E,512] x [86,512]^T + b8 -> d_out [E,86]
  {
    dim3 g((unsigned)(W3R / 128), (unsigned)(EP / 64));
    wmma_gemm_bf16<<<g, 256, 0, stream>>>(A3, w8b, (float*)d_out, b8,
                                          E, NOUT, L2LEN, NOUT);
  }
}
